// ALIGNNFF_49727131353879
// MI455X (gfx1250) — matrix operand order repaired and checked by reading, not compiled, not run on hardware
//
#include <hip/hip_runtime.h>
#include <hip/hip_bf16.h>

// ---------------------------------------------------------------------------
// ALIGNN-FF forward for MI455X (gfx1250, wave32).
//  * All linears -> v_wmma_f32_16x16x32_f16 tiled GEMM (f16 in, f32 accum).
//    One wave computes a 16-row x N block (NT=16 WMMA tiles for N=256) so the
//    activation matrix is streamed from HBM exactly once per GEMM; the f16
//    pre-transposed weight (L2-resident) slice is staged cooperatively per
//    block. No divergent guards anywhere: M % 16 == 0, N % 16 == 0 for all
//    shapes here, and K is zero-padded to a multiple of 32 at generation time.
//  * Gather/scatter message passing with native global_atomic_add_f32.
//  * Wave-per-row fused LayerNorm+SiLU epilogues (wave32 shfl reductions).
// Roofline: ~1 TFLOP of GEMM (trivial on the matrix pipes) vs ~20-25 GB of
// feature traffic + ~1.8B fp32 scatter atomics -> HBM-bound, so everything
// streams with 128b vector accesses and fused epilogues.
// ---------------------------------------------------------------------------

#define H      256
#define EMBD   64
#define EBINS  80
#define EBINSP 96     // padded K for the 80-bin RBF
#define TBINS  40
#define TBINSP 64     // padded K for the 40-bin RBF
#define NATOMC 10000
#define NEDGEC 200000
#define NTRIC  500000
#define RCUT   5.0f
#define PI_F   3.14159265358979323846f

typedef _Float16 half4 __attribute__((ext_vector_type(4)));
typedef _Float16 half8 __attribute__((ext_vector_type(8)));
typedef _Float16 v16h  __attribute__((ext_vector_type(16)));
typedef float    v8f   __attribute__((ext_vector_type(8)));

__device__ __forceinline__ float sigmoidf_(float x) { return 1.0f / (1.0f + __expf(-x)); }
__device__ __forceinline__ float siluf_(float x)    { return x * sigmoidf_(x); }

// Non-returning fp32 global atomic add (TH=0 -> no return, tracked by STOREcnt).
__device__ __forceinline__ void gatomic_add_f32(float* p, float v) {
  asm volatile("global_atomic_add_f32 %0, %1, off" :: "v"(p), "v"(v) : "memory");
}

__device__ __forceinline__ float wave_sum32(float v) {
#pragma unroll
  for (int o = 16; o > 0; o >>= 1) v += __shfl_xor(v, o, 32);
  return v;
}

// ---------------------------------------------------------------------------
// Weight pre-transpose: W fp32 [K,N] -> Wt f16 [N,Kp], zero-padded k >= K.
// Tiny (<= 64K elems), runs once per GEMM, keeps the GEMM B-path contiguous.
// ---------------------------------------------------------------------------
__global__ void wt_transpose_kernel(const float* __restrict__ W, _Float16* __restrict__ Wt,
                                    int K, int N, int Kp) {
  const long idx = (long)blockIdx.x * 256 + threadIdx.x;  // over N*Kp
  if (idx >= (long)N * Kp) return;
  const int n = (int)(idx / Kp), k = (int)(idx % Kp);
  Wt[idx] = (k < K) ? (_Float16)W[(long)k * N + n] : (_Float16)0.0f;
}

// ---------------------------------------------------------------------------
// WMMA GEMM: C[M,N] = A[M,Kp] @ Wt[N,Kp]^T + bias[N]
//   blockDim = 256 (8 waves); wave wv handles rows [m0, m0+16) x all N.
//   Per k-step: block cooperatively stages the [N,32] f16 weight slice into
//   LDS; each wave stages its own 16x32 f16 A tile; then NT back-to-back
//   v_wmma_f32_16x16x32_f16. Requires M%16==0, N==NT*16, Kp%32==0.
// Lane data layouts (CDNA5 ISA §7.12.2), g = lane>>4, l16 = lane&15:
//   A v16h: halves 0..7 -> K=g*8..g*8+7, halves 8..15 -> K=16+g*8.. (row l16)
//   B v16h: halves 0..7 -> K=g*8..g*8+7, halves 8..15 -> K=16+g*8.. (column l16)  [staged transposed]
//   D v8f:  element v -> row v+8*g, column l16
// ---------------------------------------------------------------------------
template <int NT>
__global__ __launch_bounds__(256)
void wmma_gemm_kernel(const float* __restrict__ A, const _Float16* __restrict__ Wt,
                      const float* __restrict__ bias, float* __restrict__ C,
                      int M, int Kp) {
  constexpr int N = NT * 16;
  __shared__ __align__(16) _Float16 Bs[N * 32];       // weight k-slice, shared
  __shared__ __align__(16) _Float16 As[8][16 * 32];   // per-wave A tiles
  const int tid  = threadIdx.x;
  const int lane = tid & 31;
  const int wv   = tid >> 5;
  const int m0   = (blockIdx.x * 8 + wv) * 16;
  const bool active = m0 < M;                         // wave-uniform
  const int g = lane >> 4, l16 = lane & 15;

  v8f acc[NT] = {};
  for (int k0 = 0; k0 < Kp; k0 += 32) {
    // Cooperative B stage: N*4 half8 chunks over 256 threads (contiguous 16B).
#pragma unroll
    for (int i = 0; i < NT / 4; ++i) {
      const int flat = i * 256 + tid;                 // 0 .. N*4-1
      const int n = flat >> 2, c8 = (flat & 3) * 8;
      *(half8*)&Bs[n * 32 + c8] = *(const half8*)&Wt[(long)n * Kp + k0 + c8];
    }
    // Per-wave A stage: 4 float4 per lane (128B coalesced per 8 lanes).
    if (active) {
#pragma unroll
      for (int t = 0; t < 4; ++t) {
        const int flat = t * 32 + lane;               // 0..127
        const int rrow = flat >> 3, c4 = (flat & 7) * 4;
        const float4 va = *(const float4*)&A[(long)(m0 + rrow) * Kp + k0 + c4];
        half4 h;
        h.x = (_Float16)va.x; h.y = (_Float16)va.y;
        h.z = (_Float16)va.z; h.w = (_Float16)va.w;
        *(half4*)&As[wv][rrow * 32 + c4] = h;
      }
    }
    __syncthreads();
    if (active) {
      const half8 a0 = *(const half8*)&As[wv][l16 * 32 + g * 8];
      const half8 a1 = *(const half8*)&As[wv][l16 * 32 + 16 + g * 8];
      const v16h av = __builtin_shufflevector(a0, a1, 0,1,2,3,4,5,6,7,8,9,10,11,12,13,14,15);
#pragma unroll
      for (int nt = 0; nt < NT; ++nt) {
        const half8 b0 = *(const half8*)&Bs[(nt * 16 + l16) * 32 + g * 8];
        const half8 b1 = *(const half8*)&Bs[(nt * 16 + l16) * 32 + 16 + g * 8];
        const v16h bv = __builtin_shufflevector(b0, b1, 0,1,2,3,4,5,6,7,8,9,10,11,12,13,14,15);
        acc[nt] = __builtin_amdgcn_wmma_f32_16x16x32_f16(false, av, false, bv,
                                                         (short)0, acc[nt], false, false);
      }
    }
    __syncthreads();  // protect Bs/As before next k-step restage
  }
  if (active) {
#pragma unroll
    for (int nt = 0; nt < NT; ++nt) {
      const int n = nt * 16 + l16;
      const float bb = bias[n];
#pragma unroll
      for (int v = 0; v < 8; ++v) {
        C[(long)(m0 + (g << 3) + v) * N + n] = acc[nt][v] + bb;
      }
    }
  }
}

// ---------------------------------------------------------------------------
// Fused LayerNorm + SiLU (wave per row), Y = silu(LN(X))
// ---------------------------------------------------------------------------
template <int D>
__global__ __launch_bounds__(256)
void ln_silu_kernel(const float* __restrict__ X, float* __restrict__ Y,
                    const float* __restrict__ gm, const float* __restrict__ bt,
                    int rows) {
  const int lane = threadIdx.x & 31;
  const long row = (long)blockIdx.x * 8 + (threadIdx.x >> 5);
  if (row >= rows) return;
  constexpr int P = D / 32;
  float v[P];
  float s = 0.0f;
#pragma unroll
  for (int i = 0; i < P; ++i) { v[i] = X[row * D + lane + i * 32]; s += v[i]; }
  const float mu = wave_sum32(s) / D;
  float q = 0.0f;
#pragma unroll
  for (int i = 0; i < P; ++i) { const float d = v[i] - mu; q += d * d; }
  const float rs = rsqrtf(wave_sum32(q) / D + 1e-5f);
#pragma unroll
  for (int i = 0; i < P; ++i) {
    const int j = lane + i * 32;
    const float t = gm[j] * (v[i] - mu) * rs + bt[j];
    Y[row * D + j] = siluf_(t);
  }
}

// x[row] += silu(LN(Ux[row] + ssh[row]/(ss[row]+1e-6)))
__global__ __launch_bounds__(256)
void node_update_kernel(float* __restrict__ x, const float* __restrict__ Ux,
                        const float* __restrict__ ssh, const float* __restrict__ ss,
                        const float* __restrict__ gm, const float* __restrict__ bt,
                        int rows) {
  const int lane = threadIdx.x & 31;
  const long row = (long)blockIdx.x * 8 + (threadIdx.x >> 5);
  if (row >= rows) return;
  float v[8];
  float s = 0.0f;
#pragma unroll
  for (int i = 0; i < 8; ++i) {
    const int j = lane + i * 32;
    v[i] = Ux[row * H + j] + ssh[row * H + j] / (ss[row * H + j] + 1e-6f);
    s += v[i];
  }
  const float mu = wave_sum32(s) / H;
  float q = 0.0f;
#pragma unroll
  for (int i = 0; i < 8; ++i) { const float d = v[i] - mu; q += d * d; }
  const float rs = rsqrtf(wave_sum32(q) / H + 1e-5f);
#pragma unroll
  for (int i = 0; i < 8; ++i) {
    const int j = lane + i * 32;
    const float t = gm[j] * (v[i] - mu) * rs + bt[j];
    x[row * H + j] += siluf_(t);
  }
}

// y[row] += silu(LN(m[row]))   (or silu(m[row]) when skipln)
__global__ __launch_bounds__(256)
void edge_update_kernel(float* __restrict__ y, const float* __restrict__ m,
                        const float* __restrict__ gm, const float* __restrict__ bt,
                        int rows, int skipln) {
  const int lane = threadIdx.x & 31;
  const long row = (long)blockIdx.x * 8 + (threadIdx.x >> 5);
  if (row >= rows) return;
  float v[8];
  float s = 0.0f;
#pragma unroll
  for (int i = 0; i < 8; ++i) { v[i] = m[row * H + lane + i * 32]; s += v[i]; }
  if (skipln) {
#pragma unroll
    for (int i = 0; i < 8; ++i) y[row * H + lane + i * 32] += siluf_(v[i]);
    return;
  }
  const float mu = wave_sum32(s) / H;
  float q = 0.0f;
#pragma unroll
  for (int i = 0; i < 8; ++i) { const float d = v[i] - mu; q += d * d; }
  const float rs = rsqrtf(wave_sum32(q) / H + 1e-5f);
#pragma unroll
  for (int i = 0; i < 8; ++i) {
    const int j = lane + i * 32;
    const float t = gm[j] * (v[i] - mu) * rs + bt[j];
    y[row * H + j] += siluf_(t);
  }
}

// gate[e] += AxS[src]+AxD[dst]; sigma=sigmoid(gate)*mask; atomic segment sums.
__global__ __launch_bounds__(H)
void gate_scatter_kernel(float* __restrict__ gate, const float* __restrict__ AxS,
                         const float* __restrict__ AxD, const float* __restrict__ Bh,
                         const int* __restrict__ sidx, const int* __restrict__ tidx,
                         float* __restrict__ ssh, float* __restrict__ ss,
                         const float* __restrict__ mask, int nE) {
  const long e = blockIdx.x;
  if (e >= nE) return;
  const int j = threadIdx.x;
  const long s = sidx[e], t = tidx[e];
  const float m = gate[e * H + j] + AxS[s * H + j] + AxD[t * H + j];
  gate[e * H + j] = m;                    // pre-mask gate (becomes edge update)
  float sg = sigmoidf_(m);
  if (mask) sg *= mask[e];
  gatomic_add_f32(&ssh[t * H + j], sg * Bh[s * H + j]);
  gatomic_add_f32(&ss[t * H + j], sg);
}

__global__ void embed_kernel(const int* __restrict__ an, const float* __restrict__ emb,
                             float* __restrict__ x, int n) {
  const long idx = (long)blockIdx.x * blockDim.x + threadIdx.x;
  if (idx >= (long)n * H) return;
  const long i = idx / H, j = idx % H;
  x[idx] = emb[(long)an[i] * H + j];
}

__global__ void edge_len_kernel(const float* __restrict__ r, float* __restrict__ dlen,
                                float* __restrict__ fcut, int n) {
  const int i = blockIdx.x * blockDim.x + threadIdx.x;
  if (i >= n) return;
  const float a = r[3 * i], b = r[3 * i + 1], c = r[3 * i + 2];
  const float d = sqrtf(a * a + b * b + c * c);
  dlen[i] = d;
  fcut[i] = (d < RCUT) ? 0.5f * (cosf(PI_F * d / RCUT) + 1.0f) : 0.0f;
}

// RBF rows with zero-padded stride Kp (pad columns -> 0 so padded-K GEMM is exact)
__global__ void rbf_kernel(const float* __restrict__ d, float* __restrict__ out,
                           int rows, int bins, int Kp, float vmin, float vmax) {
  const long idx = (long)blockIdx.x * blockDim.x + threadIdx.x;
  if (idx >= (long)rows * Kp) return;
  const long e = idx / Kp;
  const int b = (int)(idx % Kp);
  float val = 0.0f;
  if (b < bins) {
    const float step  = (vmax - vmin) / (bins - 1);
    const float gamma = 1.0f / step;
    const float c = vmin + b * step;
    const float t = d[e] - c;
    val = __expf(-gamma * t * t);
  }
  out[idx] = val;
}

__global__ void tri_kernel(const float* __restrict__ r, const int* __restrict__ ls,
                           const int* __restrict__ ld, const float* __restrict__ fcut,
                           float* __restrict__ cosb, float* __restrict__ tmask, int n) {
  const int i = blockIdx.x * blockDim.x + threadIdx.x;
  if (i >= n) return;
  const int a = ls[i], b = ld[i];
  const float x1 = r[3 * a], y1 = r[3 * a + 1], z1 = r[3 * a + 2];
  const float x2 = r[3 * b], y2 = r[3 * b + 1], z2 = r[3 * b + 2];
  const float dot = -(x1 * x2 + y1 * y2 + z1 * z2);   // r1 = -r[lg_src]
  const float n1 = sqrtf(x1 * x1 + y1 * y1 + z1 * z1);
  const float n2 = sqrtf(x2 * x2 + y2 * y2 + z2 * z2);
  float cv = dot / (n1 * n2);
  cv = fminf(1.0f, fmaxf(-1.0f, cv));
  cosb[i] = cv;
  tmask[i] = (fcut[a] * fcut[b] > 0.0f) ? 1.0f : 0.0f;
}

__global__ __launch_bounds__(256)
void readout_kernel(const float* __restrict__ x, const float* __restrict__ fcW,
                    const float* __restrict__ fcb, float* __restrict__ out, int n) {
  const int lane = threadIdx.x & 31;
  const long row = (long)blockIdx.x * 8 + (threadIdx.x >> 5);
  if (row >= n) return;
  float s = 0.0f;
#pragma unroll
  for (int i = 0; i < 8; ++i) { const int j = lane + i * 32; s += x[row * H + j] * fcW[j]; }
  s = wave_sum32(s);
  if (lane == 0) gatomic_add_f32(out, s + fcb[0]);
}

// ---------------------------------------------------------------------------
// Host-side orchestration
// ---------------------------------------------------------------------------
struct Lin { const float* W; const float* b; };
struct LNp { const float* g; const float* b; };
struct EGC { Lin src_gate, dst_gate, edge_gate, src_update, dst_update; LNp ln_nodes, ln_edges; };
struct MLPp { Lin l1, l2; LNp ln1, ln2; };

static void gemm(hipStream_t st, const float* A, const float* W, const float* b,
                 float* C, _Float16* WtBuf, int M, int Kreal, int Kp, int N) {
  wt_transpose_kernel<<<(int)(((long)N * Kp + 255) / 256), 256, 0, st>>>(W, WtBuf, Kreal, N, Kp);
  const int blocks = (M + 127) / 128;
  if (N == 256) wmma_gemm_kernel<16><<<blocks, 256, 0, st>>>(A, WtBuf, b, C, M, Kp);
  else          wmma_gemm_kernel<4><<<blocks, 256, 0, st>>>(A, WtBuf, b, C, M, Kp);
}

static void run_egc(hipStream_t st, const EGC& p, const int* s, const int* t,
                    float* x, float* y, int xr, int er,
                    float* gate, float* AxS, float* AxD, float* Bh, float* Ux,
                    float* ssh, float* ss, _Float16* wt, const float* mask, bool skipln) {
  gemm(st, x, p.src_gate.W,   p.src_gate.b,   AxS,  wt, xr, H, H, H);
  gemm(st, x, p.dst_gate.W,   p.dst_gate.b,   AxD,  wt, xr, H, H, H);
  gemm(st, x, p.dst_update.W, p.dst_update.b, Bh,   wt, xr, H, H, H);
  gemm(st, x, p.src_update.W, p.src_update.b, Ux,   wt, xr, H, H, H);
  gemm(st, y, p.edge_gate.W,  p.edge_gate.b,  gate, wt, er, H, H, H);
  hipMemsetAsync(ssh, 0, (size_t)xr * H * sizeof(float), st);
  hipMemsetAsync(ss,  0, (size_t)xr * H * sizeof(float), st);
  gate_scatter_kernel<<<er, H, 0, st>>>(gate, AxS, AxD, Bh, s, t, ssh, ss, mask, er);
  node_update_kernel<<<(xr + 7) / 8, 256, 0, st>>>(x, Ux, ssh, ss,
                                                   p.ln_nodes.g, p.ln_nodes.b, xr);
  edge_update_kernel<<<(er + 7) / 8, 256, 0, st>>>(y, gate,
                                                   p.ln_edges.g, p.ln_edges.b, er,
                                                   skipln ? 1 : 0);
}

extern "C" void kernel_launch(void* const* d_in, const int* in_sizes, int n_in,
                              void* d_out, int out_size, void* d_ws, size_t ws_size,
                              hipStream_t stream) {
  (void)in_sizes; (void)n_in; (void)out_size; (void)ws_size;
  // Graph arrays (setup_inputs dict order).
  const int*   an     = (const int*)d_in[0];
  const float* r      = (const float*)d_in[1];
  const int*   src    = (const int*)d_in[2];
  const int*   dst    = (const int*)d_in[3];
  const int*   lg_src = (const int*)d_in[4];
  const int*   lg_dst = (const int*)d_in[5];

  // Params: JAX pytree flatten order (sorted dict keys).
  // params: alignn(4x{edge,node} EGC) -> angle_emb(MLP) -> atom_emb -> edge_emb(MLP) -> fc -> gcn(4x EGC)
  // EGC leaves (14): dst_gate.{W,b} dst_update.{W,b} edge_gate.{W,b}
  //                  ln_edges.{b,g} ln_nodes.{b,g} src_gate.{W,b} src_update.{W,b}
  // MLP leaves (8):  l1.{W,b} l2.{W,b} ln1.{b,g} ln2.{b,g}
  auto leaf = [&](int i) -> const float* { return (const float*)d_in[6 + i]; };
  auto get_egc = [&](int base) -> EGC {
    EGC e;
    e.dst_gate   = { leaf(base + 0),  leaf(base + 1)  };
    e.dst_update = { leaf(base + 2),  leaf(base + 3)  };
    e.edge_gate  = { leaf(base + 4),  leaf(base + 5)  };
    e.ln_edges   = { leaf(base + 7),  leaf(base + 6)  };   // {g,b}
    e.ln_nodes   = { leaf(base + 9),  leaf(base + 8)  };
    e.src_gate   = { leaf(base + 10), leaf(base + 11) };
    e.src_update = { leaf(base + 12), leaf(base + 13) };
    return e;
  };
  auto get_mlp = [&](int base) -> MLPp {
    MLPp m;
    m.l1  = { leaf(base + 0), leaf(base + 1) };
    m.l2  = { leaf(base + 2), leaf(base + 3) };
    m.ln1 = { leaf(base + 5), leaf(base + 4) };
    m.ln2 = { leaf(base + 7), leaf(base + 6) };
    return m;
  };
  EGC alignn_edge[4], alignn_node[4], gcn[4];
  for (int i = 0; i < 4; ++i) {
    alignn_edge[i] = get_egc(i * 28);
    alignn_node[i] = get_egc(i * 28 + 14);
  }
  const MLPp  angle_mlp = get_mlp(112);
  const float* atom_emb = leaf(120);
  const MLPp  edge_mlp  = get_mlp(121);
  const float* fcW = leaf(129);
  const float* fcb = leaf(130);
  for (int i = 0; i < 4; ++i) gcn[i] = get_egc(131 + i * 14);

  // Workspace carve (256B aligned). Total ~2.5 GB.
  char* wp = (char*)d_ws;
  auto carve = [&](size_t bytes) -> char* {
    char* q = wp;
    wp += (bytes + 255) & ~(size_t)255;
    return q;
  };
  float* x    = (float*)carve((size_t)NATOMC * H * 4);
  float* y    = (float*)carve((size_t)NEDGEC * H * 4);
  float* z    = (float*)carve((size_t)NTRIC  * H * 4);
  float* gate = (float*)carve((size_t)NTRIC  * H * 4);   // also RBF scratch
  float* bufS = (float*)carve((size_t)NEDGEC * H * 4);   // also [rows,64] MLP hidden
  float* bufD = (float*)carve((size_t)NEDGEC * H * 4);
  float* bufB = (float*)carve((size_t)NEDGEC * H * 4);
  float* bufU = (float*)carve((size_t)NEDGEC * H * 4);
  float* ssh  = (float*)carve((size_t)NEDGEC * H * 4);
  float* ssum = (float*)carve((size_t)NEDGEC * H * 4);
  float* dlen = (float*)carve((size_t)NEDGEC * 4);
  float* fcut = (float*)carve((size_t)NEDGEC * 4);
  float* cosb = (float*)carve((size_t)NTRIC * 4);
  float* tmsk = (float*)carve((size_t)NTRIC * 4);
  _Float16* wt = (_Float16*)carve((size_t)H * H * 2);    // f16 transposed weight

  float* out = (float*)d_out;
  hipMemsetAsync(out, 0, sizeof(float), stream);

  // 1) Atom embeddings.
  embed_kernel<<<(NATOMC * H + 255) / 256, 256, 0, stream>>>(an, atom_emb, x, NATOMC);

  // 2) Edge embeddings: d -> RBF(80, pad 96) -> Lin/LN/SiLU -> Lin/LN/SiLU -> y.
  edge_len_kernel<<<(NEDGEC + 255) / 256, 256, 0, stream>>>(r, dlen, fcut, NEDGEC);
  rbf_kernel<<<(int)(((long)NEDGEC * EBINSP + 255) / 256), 256, 0, stream>>>(
      dlen, gate, NEDGEC, EBINS, EBINSP, 0.0f, 8.0f);
  gemm(stream, gate, edge_mlp.l1.W, edge_mlp.l1.b, bufS, wt, NEDGEC, EBINS, EBINSP, EMBD);
  ln_silu_kernel<EMBD><<<(NEDGEC + 7) / 8, 256, 0, stream>>>(bufS, bufS, edge_mlp.ln1.g, edge_mlp.ln1.b, NEDGEC);
  gemm(stream, bufS, edge_mlp.l2.W, edge_mlp.l2.b, y, wt, NEDGEC, EMBD, EMBD, H);
  ln_silu_kernel<H><<<(NEDGEC + 7) / 8, 256, 0, stream>>>(y, y, edge_mlp.ln2.g, edge_mlp.ln2.b, NEDGEC);

  // 3) Angle embeddings: cos -> RBF(40, pad 64) -> MLP -> z ; triplet mask.
  tri_kernel<<<(NTRIC + 255) / 256, 256, 0, stream>>>(r, lg_src, lg_dst, fcut, cosb, tmsk, NTRIC);
  rbf_kernel<<<(int)(((long)NTRIC * TBINSP + 255) / 256), 256, 0, stream>>>(
      cosb, gate, NTRIC, TBINS, TBINSP, -1.0f, 1.0f);
  gemm(stream, gate, angle_mlp.l1.W, angle_mlp.l1.b, bufS, wt, NTRIC, TBINS, TBINSP, EMBD);
  ln_silu_kernel<EMBD><<<(NTRIC + 7) / 8, 256, 0, stream>>>(bufS, bufS, angle_mlp.ln1.g, angle_mlp.ln1.b, NTRIC);
  gemm(stream, bufS, angle_mlp.l2.W, angle_mlp.l2.b, z, wt, NTRIC, EMBD, EMBD, H);
  ln_silu_kernel<H><<<(NTRIC + 7) / 8, 256, 0, stream>>>(z, z, angle_mlp.ln2.g, angle_mlp.ln2.b, NTRIC);

  // 4) ALIGNN layers: atom-graph conv then line-graph conv (in-place updates
  //    reproduce the reference's (x, m=y_new) -> (y_new, z_new) dataflow).
  for (int i = 0; i < 4; ++i) {
    run_egc(stream, alignn_node[i], src, dst, x, y, NATOMC, NEDGEC,
            gate, bufS, bufD, bufB, bufU, ssh, ssum, wt, nullptr, false);
    run_egc(stream, alignn_edge[i], lg_src, lg_dst, y, z, NEDGEC, NTRIC,
            gate, bufS, bufD, bufB, bufU, ssh, ssum, wt, tmsk, i == 3);
  }
  // 5) GCN layers.
  for (int i = 0; i < 4; ++i) {
    run_egc(stream, gcn[i], src, dst, x, y, NATOMC, NEDGEC,
            gate, bufS, bufD, bufB, bufU, ssh, ssum, wt, nullptr, i == 3);
  }

  // 6) Readout: sum over atoms of (x @ fcW + fcb).
  readout_kernel<<<(NATOMC + 7) / 8, 256, 0, stream>>>(x, fcW, fcb, out, NATOMC);
}
